// RefLocal7MH_21689584845203
// MI455X (gfx1250) — matrix operand order repaired and checked by reading, not compiled, not run on hardware
//
#include <hip/hip_runtime.h>
#include <hip/hip_bf16.h>

typedef __attribute__((ext_vector_type(16))) __bf16 v16bf;
typedef __attribute__((ext_vector_type(8)))  __bf16 v8bf;
typedef __attribute__((ext_vector_type(8)))  float  v8f;
typedef __attribute__((ext_vector_type(4)))  float  v4f;

#define IMG_H 128
#define IMG_W 128
#define CH    64
#define TILE  16
#define HALO  20          // TILE + 4
#define NPIX  256         // TILE*TILE
#define NREF  400         // HALO*HALO
#define PITCH_BF 72       // bf16 row pitch (144B, 16B-aligned, bank-conflict-free)
#define PITCH_F  68       // f32 row pitch (272B, 16B-aligned, bank-conflict-free)
#define NKEY  26

// ---- WMMA fragment loaders (CDNA5 ISA 7.12.2 layouts) -----------------------
// A (16x32 bf16): lane<16 row M=lane holds K {0..7, 16..23}; lane>=16 holds
// K {8..15, 24..31}; VGPR0-3 = first run, VGPR4-7 = second run (2 bf16/VGPR).
__device__ inline v16bf load_fragA(const __bf16* buf, int row, int ks, int laneHi) {
    const __bf16* p = buf + row * PITCH_BF + ks * 32 + laneHi * 8;
    v8bf lo = *(const v8bf*)p;          // ds_load_b128
    v8bf hi = *(const v8bf*)(p + 16);   // ds_load_b128
    v16bf a;
#pragma unroll
    for (int i = 0; i < 8; ++i) { a[i] = lo[i]; a[i + 8] = hi[i]; }
    return a;
}
// B (32x16 bf16): lane<16 col N=lane holds K {0..7, 16..23}; lane>=16 holds
// K {8..15, 24..31}. Weights staged transposed as Wt[out][c] so runs are contiguous.
__device__ inline v16bf load_fragB(const __bf16* wt, int ncol, int ks, int laneHi) {
    const __bf16* p = wt + ncol * PITCH_BF + ks * 32 + laneHi * 8;
    v8bf lo = *(const v8bf*)p;
    v8bf hi = *(const v8bf*)(p + 16);
    v16bf b;
#pragma unroll
    for (int i = 0; i < 8; ++i) { b[i] = lo[i]; b[i + 8] = hi[i]; }
    return b;
}

__launch_bounds__(256, 1)
__global__ void local_attn_fused(const float* __restrict__ main_in,
                                 const float* __restrict__ ref_in,
                                 const float* __restrict__ Wm,
                                 const float* __restrict__ Wr,
                                 float* __restrict__ out)
{
    __shared__ __bf16 sWt[2][64 * PITCH_BF];      // transposed weights [out][c]
    __shared__ __bf16 sIn[NREF * PITCH_BF];       // staged inputs (main, then ref)
    __shared__ float  sQ[NPIX * PITCH_F];         // projected queries, f32
    __shared__ float  sK[NREF * PITCH_F];         // projected keys (halo), f32

    const int tid  = threadIdx.x;
    const int lane = tid & 31;
    const int wv   = tid >> 5;            // wave id, 0..7
    const int laneHi = lane >> 4;
    const int lane16 = lane & 15;

    const int bx = blockIdx.x;
    const int b  = bx >> 6;
    const int ty = (bx >> 3) & 7;
    const int tx = bx & 7;
    const int h0 = ty * TILE;
    const int w0 = tx * TILE;
    const long imgBase = (long)b * IMG_H * IMG_W * CH;

    // ---- Phase 0: stage transposed bf16 weights: Wt[out=n*16+d][c] ----------
    for (int e = tid; e < 64 * 64; e += 256) {
        int o = e >> 6, c = e & 63;
        int n = o >> 4, d = o & 15;
        int src = n * (64 * 16) + c * 16 + d;
        sWt[0][o * PITCH_BF + c] = (__bf16)Wm[src];
        sWt[1][o * PITCH_BF + c] = (__bf16)Wr[src];
    }

    // ---- Phase 1: stage main tile (256 px x 64 ch), coalesced ---------------
    for (int e = tid; e < NPIX * CH; e += 256) {
        int px = e >> 6, ch = e & 63;
        int gy = h0 + (px >> 4), gx = w0 + (px & 15);
        sIn[px * PITCH_BF + ch] =
            (__bf16)main_in[imgBase + ((long)gy * IMG_W + gx) * CH + ch];
    }
    __syncthreads();

    // ---- Phase 2a: Q = mainTile(256x64) @ Wm(64x64) on the matrix pipe ------
    for (int t = wv; t < 64; t += 8) {          // 16 m-tiles x 4 n-tiles
        int mt = t >> 2, nt = t & 3;
        v8f acc = {};
#pragma unroll
        for (int ks = 0; ks < 2; ++ks) {
            v16bf a  = load_fragA(sIn, mt * 16 + lane16, ks, laneHi);
            v16bf bq = load_fragB(sWt[0], nt * 16 + lane16, ks, laneHi);
            acc = __builtin_amdgcn_wmma_f32_16x16x32_bf16(
                false, a, false, bq, (short)0, acc, false, false);
        }
        int r0 = mt * 16 + laneHi * 8;          // D layout: VGPR r <-> row r(+8)
#pragma unroll
        for (int r = 0; r < 8; ++r)
            sQ[(r0 + r) * PITCH_F + nt * 16 + lane16] = acc[r];
    }
    __syncthreads();    // sIn about to be overwritten

    // ---- Phase 1b: stage ref halo (20x20 px, zero-padded SAME) --------------
    for (int e = tid; e < NREF * CH; e += 256) {
        int px = e >> 6, ch = e & 63;
        int hy = px / HALO, hx = px % HALO;
        int gy = h0 + hy - 2, gx = w0 + hx - 2;
        float v = 0.f;
        if (gy >= 0 && gy < IMG_H && gx >= 0 && gx < IMG_W)
            v = ref_in[imgBase + ((long)gy * IMG_W + gx) * CH + ch];
        sIn[px * PITCH_BF + ch] = (__bf16)v;
    }
    __syncthreads();

    // ---- Phase 2b: K = refHalo(400x64) @ Wr(64x64) --------------------------
    for (int t = wv; t < 100; t += 8) {         // 25 m-tiles x 4 n-tiles
        int mt = t >> 2, nt = t & 3;
        v8f acc = {};
#pragma unroll
        for (int ks = 0; ks < 2; ++ks) {
            v16bf a  = load_fragA(sIn, mt * 16 + lane16, ks, laneHi);
            v16bf bk = load_fragB(sWt[1], nt * 16 + lane16, ks, laneHi);
            acc = __builtin_amdgcn_wmma_f32_16x16x32_bf16(
                false, a, false, bk, (short)0, acc, false, false);
        }
        int r0 = mt * 16 + laneHi * 8;
#pragma unroll
        for (int r = 0; r < 8; ++r)
            sK[(r0 + r) * PITCH_F + nt * 16 + lane16] = acc[r];
    }
    __syncthreads();

    // ---- Phase 3: per-pixel 26-way scores, softmax, head-average ------------
    const int ly = tid >> 4, lx = tid & 15;
    float outAcc[NKEY];
#pragma unroll
    for (int j = 0; j < NKEY; ++j) outAcc[j] = 0.f;

    for (int n = 0; n < 4; ++n) {
        float q[16];
        const float* qp = &sQ[tid * PITCH_F + n * 16];
#pragma unroll
        for (int i = 0; i < 16; i += 4) {
            v4f t4 = *(const v4f*)(qp + i);
            q[i] = t4.x; q[i + 1] = t4.y; q[i + 2] = t4.z; q[i + 3] = t4.w;
        }
        float sc[NKEY];
        int idx = 0;
#pragma unroll
        for (int di = 0; di < 5; ++di) {
#pragma unroll
            for (int dj = 0; dj < 5; ++dj) {
                const float* kp = &sK[((ly + di) * HALO + (lx + dj)) * PITCH_F + n * 16];
                float s = 0.f;
#pragma unroll
                for (int i = 0; i < 16; i += 4) {
                    v4f k4 = *(const v4f*)(kp + i);
                    s += q[i] * k4.x + q[i + 1] * k4.y + q[i + 2] * k4.z + q[i + 3] * k4.w;
                }
                sc[idx++] = s;
            }
        }
        float s25 = 0.f;
#pragma unroll
        for (int i = 0; i < 16; ++i) s25 += q[i] * q[i];
        sc[25] = s25;

        float m = sc[0];
#pragma unroll
        for (int j = 1; j < NKEY; ++j) m = fmaxf(m, sc[j]);
        float sum = 0.f;
#pragma unroll
        for (int j = 0; j < NKEY; ++j) { sc[j] = __expf(sc[j] - m); sum += sc[j]; }
        float inv = 0.25f / sum;   // mean over 4 heads folded in
#pragma unroll
        for (int j = 0; j < NKEY; ++j) outAcc[j] += sc[j] * inv;
    }

    const long obase = (((long)b * IMG_H + (h0 + ly)) * IMG_W + (w0 + lx)) * NKEY;
#pragma unroll
    for (int j = 0; j < NKEY; ++j) out[obase + j] = outAcc[j];
}

extern "C" void kernel_launch(void* const* d_in, const int* in_sizes, int n_in,
                              void* d_out, int out_size, void* d_ws, size_t ws_size,
                              hipStream_t stream) {
    const float* main_in = (const float*)d_in[0];
    const float* ref_in  = (const float*)d_in[1];
    const float* Wm      = (const float*)d_in[2];
    const float* Wr      = (const float*)d_in[3];
    float* out = (float*)d_out;

    dim3 grid(8 * 8 * 8);   // B * (H/16) * (W/16)
    dim3 block(256);
    local_attn_fused<<<grid, block, 0, stream>>>(main_in, ref_in, Wm, Wr, out);
}
